// SelfAttention_2138893714151
// MI455X (gfx1250) — matrix operand order repaired and checked by reading, not compiled, not run on hardware
//
#include <hip/hip_runtime.h>
#include <hip/hip_bf16.h>

#define EMBED 2048
#define NHEADS 16
#define HDIM 128
#define BB 2
#define TT 2048
#define BK 32
#define LDS_STRIDE 40   // 32 data halves + 8 pad halves (80B rows -> conflict-free b128 frags)

typedef _Float16 half8  __attribute__((ext_vector_type(8)));
typedef _Float16 v16h   __attribute__((ext_vector_type(16)));
typedef float    v8f    __attribute__((ext_vector_type(8)));
typedef unsigned int u32x4 __attribute__((ext_vector_type(4)));
typedef int      i32x4 __attribute__((ext_vector_type(4)));
typedef int      i32x8 __attribute__((ext_vector_type(8)));

#if __has_builtin(__builtin_amdgcn_tensor_load_to_lds)
#define USE_TDM 1
#else
#define USE_TDM 0
#endif

static __device__ __forceinline__ v16h ld_frag2(const _Float16* p0, const _Float16* p1) {
    union { v16h v; half8 h[2]; } u;
    u.h[0] = *reinterpret_cast<const half8*>(p0);
    u.h[1] = *reinterpret_cast<const half8*>(p1);
    return u.v;
}

static __device__ __forceinline__ v8f wmma_f16(v16h a, v16h b, v8f c) {
    // D = A(16x32,f16) * B(32x16,f16) + C(16x16,f32)
    return __builtin_amdgcn_wmma_f32_16x16x32_f16(false, a, false, b, (short)0, c, false, false);
}

#if USE_TDM
// One TDM descriptor: 2D tile 32(K) x 128(rows), f16, row pitch = rowStride elems,
// padded in LDS with 16B after every 64B row chunk (-> LDS_STRIDE halves per row).
static __device__ __forceinline__ void tdm_load_tile(_Float16* ldsDst, const _Float16* gSrc,
                                                     int rowStride) {
    unsigned long long ga = (unsigned long long)(size_t)gSrc;
    u32x4 g0;
    g0.x = 1u;                                            // count=1, user descriptor
    g0.y = (unsigned)(size_t)ldsDst;                      // LDS byte address (low 32 bits)
    g0.z = (unsigned)(ga & 0xFFFFFFFFull);                // global_addr[31:0]
    g0.w = (unsigned)((ga >> 32) & 0x1FFFFFFull) | (2u << 30);  // global_addr[56:32] | type=2

    i32x8 g1;
    g1[0] = (1 << 16)        // data_size = 2B
          | (1 << 20)        // pad_enable
          | (3 << 22)        // pad_interval: 16 DWORDs (one 64B tile row)
          | (3 << 25);       // pad_amount: 4 DWORDs (16B)
    g1[1] = (rowStride & 0xFFFF) << 16;   // tensor_dim0[15:0]
    g1[2] = (int)0xFFFF0000;              // tensor_dim1[15:0]=0xFFFF (big; no OOB clip)
    g1[3] = (BK << 16);                   // tile_dim0 = 32
    g1[4] = 128;                          // tile_dim1 = 128 rows, tile_dim2 = 0
    g1[5] = rowStride;                    // tensor_dim0_stride[31:0] (row pitch)
    g1[6] = 0;
    g1[7] = 0;

    i32x4 z4 = {0, 0, 0, 0};
#if defined(__clang_major__) && __clang_major__ >= 23
    i32x8 z8 = {0, 0, 0, 0, 0, 0, 0, 0};
    __builtin_amdgcn_tensor_load_to_lds(g0, g1, z4, z4, z8, 0);
#else
    __builtin_amdgcn_tensor_load_to_lds(g0, g1, z4, z4, 0);
#endif
}
#endif

// ---------------------------------------------------------------------------
// f32 -> f16 convert
// ---------------------------------------------------------------------------
__global__ void cvt_f32_f16(const float* __restrict__ in, _Float16* __restrict__ out, int n) {
    int i = blockIdx.x * 256 + threadIdx.x;
    if (i < n) out[i] = (_Float16)in[i];
}

// ---------------------------------------------------------------------------
// GEMM: out[M][N] = A[M][K] @ W[N][K]^T + bias[N]
// block = 256 thr (8 waves). Block tile 128x128, wave tile 64(M) x 32(N).
// Tiles staged in LDS (double-buffered) via TDM (tensor_load_to_lds) when
// available, else manual global->VGPR->ds_store.
// ---------------------------------------------------------------------------
template<bool F16OUT>
__global__ void __launch_bounds__(256) gemm_bias(const _Float16* __restrict__ A,
                                                 const _Float16* __restrict__ W,
                                                 const float*    __restrict__ bias,
                                                 void*           __restrict__ outp,
                                                 int N, int K) {
    __shared__ _Float16 Ash[2][128 * LDS_STRIDE];
    __shared__ _Float16 Bsh[2][128 * LDS_STRIDE];

    const int tid  = threadIdx.x;
    const int lane = tid & 31;
    const int w    = tid >> 5;
    const int wm   = w >> 2, wn = w & 3;
    const int l15  = lane & 15;
    const int hi   = (lane & 16) ? 1 : 0;
    const int mBlk = blockIdx.y * 128;
    const int nBlk = blockIdx.x * 128;

    v8f acc[4][2];
    #pragma unroll
    for (int mt = 0; mt < 4; ++mt)
        #pragma unroll
        for (int nt = 0; nt < 2; ++nt)
            #pragma unroll
            for (int e = 0; e < 8; ++e) acc[mt][nt][e] = 0.0f;

    // ---- stage k-chunk 0 into buffer 0 ----
#if USE_TDM
    if (tid < 32) {
        tdm_load_tile(&Ash[0][0], A + (size_t)mBlk * K, K);
        tdm_load_tile(&Bsh[0][0], W + (size_t)nBlk * K, K);
    }
#else
    #pragma unroll
    for (int c = 0; c < 2; ++c) {
        const int chunk = tid + c * 256;          // 512 chunks of 8 halves per tile
        const int row = chunk >> 2;
        const int col = (chunk & 3) * 8;
        *reinterpret_cast<half8*>(&Ash[0][row * LDS_STRIDE + col]) =
            *reinterpret_cast<const half8*>(A + (size_t)(mBlk + row) * K + col);
        *reinterpret_cast<half8*>(&Bsh[0][row * LDS_STRIDE + col]) =
            *reinterpret_cast<const half8*>(W + (size_t)(nBlk + row) * K + col);
    }
#endif

    for (int kk = 0; kk < K; kk += BK) {
        const int buf = (kk >> 5) & 1;

        // ---- make current buffer visible to all waves ----
#if USE_TDM
        if (tid < 32) __builtin_amdgcn_s_wait_tensorcnt(0);
#endif
        __syncthreads();

        // ---- kick next stage into the other buffer ----
        if (kk + BK < K) {
            const int nk = kk + BK;
#if USE_TDM
            if (tid < 32) {
                tdm_load_tile(&Ash[buf ^ 1][0], A + (size_t)mBlk * K + nk, K);
                tdm_load_tile(&Bsh[buf ^ 1][0], W + (size_t)nBlk * K + nk, K);
            }
#else
            #pragma unroll
            for (int c = 0; c < 2; ++c) {
                const int chunk = tid + c * 256;
                const int row = chunk >> 2;
                const int col = (chunk & 3) * 8;
                *reinterpret_cast<half8*>(&Ash[buf ^ 1][row * LDS_STRIDE + col]) =
                    *reinterpret_cast<const half8*>(A + (size_t)(mBlk + row) * K + nk + col);
                *reinterpret_cast<half8*>(&Bsh[buf ^ 1][row * LDS_STRIDE + col]) =
                    *reinterpret_cast<const half8*>(W + (size_t)(nBlk + row) * K + nk + col);
            }
#endif
        }

        // ---- fragments from LDS, 8 WMMA ----
        v16h af[4], bf[2];
        #pragma unroll
        for (int mt = 0; mt < 4; ++mt) {
            const _Float16* p = &Ash[buf][(wm * 64 + mt * 16 + l15) * LDS_STRIDE + hi * 8];
            af[mt] = ld_frag2(p, p + 16);
        }
        #pragma unroll
        for (int nt = 0; nt < 2; ++nt) {
            const _Float16* p = &Bsh[buf][(wn * 32 + nt * 16 + l15) * LDS_STRIDE + hi * 8];
            bf[nt] = ld_frag2(p, p + 16);
        }
        #pragma unroll
        for (int mt = 0; mt < 4; ++mt)
            #pragma unroll
            for (int nt = 0; nt < 2; ++nt)
                acc[mt][nt] = wmma_f16(af[mt], bf[nt], acc[mt][nt]);

        __syncthreads();   // compute(buf) done before buf is overwritten next round
    }

    #pragma unroll
    for (int nt = 0; nt < 2; ++nt) {
        const int col = nBlk + wn * 32 + nt * 16 + l15;
        const float bv = bias[col];
        #pragma unroll
        for (int mt = 0; mt < 4; ++mt) {
            #pragma unroll
            for (int e = 0; e < 8; ++e) {
                const int row = mBlk + wm * 64 + mt * 16 + e + hi * 8;
                const float v = acc[mt][nt][e] + bv;
                if (F16OUT) ((_Float16*)outp)[(size_t)row * N + col] = (_Float16)v;
                else        ((float*)outp)[(size_t)row * N + col]    = v;
            }
        }
    }
}

// ---------------------------------------------------------------------------
// RoPE + repack: qkv[tok][6144] -> Qr/Kr [b,h,t,128], Vt [b,h,128,t]
// ---------------------------------------------------------------------------
__global__ void rope_pack(const _Float16* __restrict__ qkv,
                          _Float16* __restrict__ Qr,
                          _Float16* __restrict__ Kr,
                          _Float16* __restrict__ Vt) {
    int idx = blockIdx.x * 256 + threadIdx.x;          // 2*16*2048*64 total
    int i = idx & 63;
    int t = (idx >> 6) & 2047;
    int h = (idx >> 17) & 15;
    int b = (idx >> 21) & 1;

    size_t tok = (size_t)b * TT + t;
    const _Float16* row = qkv + tok * (3 * EMBED);
    int cb = h * HDIM;

    float freq = __powf(10000.0f, -(float)i * (1.0f / 64.0f));
    float th = (float)t * freq;
    float sn, cs;
    __sincosf(th, &sn, &cs);

    size_t obase = ((size_t)(b * NHEADS + h) * TT + t) * HDIM;
    float q1 = (float)row[cb + 2 * i], q2 = (float)row[cb + 2 * i + 1];
    Qr[obase + i]      = (_Float16)(q1 * cs - q2 * sn);
    Qr[obase + 64 + i] = (_Float16)(q1 * sn + q2 * cs);
    float k1 = (float)row[EMBED + cb + 2 * i], k2 = (float)row[EMBED + cb + 2 * i + 1];
    Kr[obase + i]      = (_Float16)(k1 * cs - k2 * sn);
    Kr[obase + 64 + i] = (_Float16)(k1 * sn + k2 * cs);

    size_t vbase = ((size_t)(b * NHEADS + h) * HDIM) * (size_t)TT + t;
    Vt[vbase + (size_t)i * TT]        = row[2 * EMBED + cb + i];
    Vt[vbase + (size_t)(64 + i) * TT] = row[2 * EMBED + cb + 64 + i];
}

// ---------------------------------------------------------------------------
// Causal flash attention. grid = (T/64, B*H), block = 128 thr (4 waves).
// ---------------------------------------------------------------------------
__global__ void __launch_bounds__(128) attn_flash(const _Float16* __restrict__ Q,
                                                  const _Float16* __restrict__ Km,
                                                  const _Float16* __restrict__ Vt,
                                                  _Float16* __restrict__ Oh) {
    __shared__ _Float16 Plds[4 * 16 * 32];             // 1KB per wave

    const int lane = threadIdx.x & 31;
    const int w    = threadIdx.x >> 5;
    const int l15  = lane & 15;
    const int hi   = (lane & 16) ? 1 : 0;

    const int bh = blockIdx.y;
    const int b  = bh >> 4;
    const int h  = bh & 15;
    const _Float16* Qb = Q  + (size_t)bh * TT * HDIM;
    const _Float16* Kb = Km + (size_t)bh * TT * HDIM;
    const _Float16* Vb = Vt + (size_t)bh * HDIM * TT;

    const int q0 = blockIdx.x * 64 + w * 16;

    v16h qf[4];
    {
        const size_t qr = (size_t)(q0 + l15) * HDIM;
        #pragma unroll
        for (int d = 0; d < 4; ++d) {
            const _Float16* p = Qb + qr + d * 32 + hi * 8;
            qf[d] = ld_frag2(p, p + 16);
        }
    }

    float mrow[8], lrow[8];
    v8f oacc[8];
    #pragma unroll
    for (int e = 0; e < 8; ++e) { mrow[e] = -1e30f; lrow[e] = 0.0f; }
    #pragma unroll
    for (int t8 = 0; t8 < 8; ++t8)
        #pragma unroll
        for (int e = 0; e < 8; ++e) oacc[t8][e] = 0.0f;

    const float scale = 0.08838834764831845f;          // 1/sqrt(128)
    const int kend = blockIdx.x * 64 + 64;

    for (int kb = 0; kb < kend; kb += 32) {
        v8f s0, s1;
        #pragma unroll
        for (int e = 0; e < 8; ++e) { s0[e] = 0.0f; s1[e] = 0.0f; }
        #pragma unroll
        for (int d = 0; d < 4; ++d) {
            const int bc = d * 32 + hi * 8;
            const _Float16* p0 = Kb + (size_t)(kb + l15) * HDIM + bc;
            const _Float16* p1 = Kb + (size_t)(kb + 16 + l15) * HDIM + bc;
            v16h b0 = ld_frag2(p0, p0 + 16);
            v16h b1 = ld_frag2(p1, p1 + 16);
            s0 = wmma_f16(qf[d], b0, s0);
            s1 = wmma_f16(qf[d], b1, s1);
        }

        const int mbase = q0 + hi * 8;
        const int jc = kb + l15;
        float p0v[8], p1v[8], corr[8];
        #pragma unroll
        for (int e = 0; e < 8; ++e) {
            const int qv = mbase + e;
            float a0 = s0[e] * scale;
            float a1 = s1[e] * scale;
            if (jc > qv)      a0 = -1e30f;
            if (jc + 16 > qv) a1 = -1e30f;
            float t = fmaxf(a0, a1);
            #pragma unroll
            for (int mm = 1; mm < 16; mm <<= 1) t = fmaxf(t, __shfl_xor(t, mm, 32));
            const float mn = fmaxf(mrow[e], t);
            const float c  = __expf(mrow[e] - mn);
            const float e0 = __expf(a0 - mn);
            const float e1 = __expf(a1 - mn);
            float rs = e0 + e1;
            #pragma unroll
            for (int mm = 1; mm < 16; mm <<= 1) rs += __shfl_xor(rs, mm, 32);
            lrow[e] = lrow[e] * c + rs;
            mrow[e] = mn;
            corr[e] = c;
            p0v[e] = e0;
            p1v[e] = e1;
        }
        #pragma unroll
        for (int t8 = 0; t8 < 8; ++t8)
            #pragma unroll
            for (int e = 0; e < 8; ++e) oacc[t8][e] *= corr[e];

        _Float16* pl = &Plds[w * 512];
        #pragma unroll
        for (int e = 0; e < 8; ++e) {
            const int m = e + hi * 8;
            pl[m * 32 + l15]      = (_Float16)p0v[e];
            pl[m * 32 + l15 + 16] = (_Float16)p1v[e];
        }
        const _Float16* pa = pl + l15 * 32 + hi * 8;
        v16h pf = ld_frag2(pa, pa + 16);

        #pragma unroll
        for (int t8 = 0; t8 < 8; ++t8) {
            const _Float16* pv = Vb + (size_t)(t8 * 16 + l15) * TT + kb + hi * 8;
            v16h vf = ld_frag2(pv, pv + 16);
            oacc[t8] = wmma_f16(pf, vf, oacc[t8]);
        }
    }

    #pragma unroll
    for (int t8 = 0; t8 < 8; ++t8) {
        #pragma unroll
        for (int e = 0; e < 8; ++e) {
            const int qrow = q0 + e + hi * 8;
            const float v = oacc[t8][e] / lrow[e];
            Oh[(size_t)(b * TT + qrow) * EMBED + h * HDIM + t8 * 16 + l15] = (_Float16)v;
        }
    }
}

// ---------------------------------------------------------------------------
// launch
// ---------------------------------------------------------------------------
extern "C" void kernel_launch(void* const* d_in, const int* in_sizes, int n_in,
                              void* d_out, int out_size, void* d_ws, size_t ws_size,
                              hipStream_t stream) {
    const float* x     = (const float*)d_in[0];   // [2,2048,2048]
    const float* w_qkv = (const float*)d_in[1];   // [6144,2048]
    const float* b_qkv = (const float*)d_in[2];   // [6144]
    const float* w_out = (const float*)d_in[3];   // [2048,2048]
    const float* b_out = (const float*)d_in[4];   // [2048]
    float* out = (float*)d_out;                   // [2,2048,2048]

    _Float16* wsh  = (_Float16*)d_ws;
    _Float16* Xh   = wsh;                  // 4096*2048
    _Float16* Wqh  = Xh   + 8388608;       // 6144*2048
    _Float16* Woh  = Wqh  + 12582912;      // 2048*2048
    _Float16* QKVh = Woh  + 4194304;       // 4096*6144
    _Float16* Qr   = QKVh + 25165824;      // 2*16*2048*128
    _Float16* Kr   = Qr   + 8388608;
    _Float16* Vt   = Kr   + 8388608;
    _Float16* Oh   = Vt   + 8388608;       // 4096*2048

    cvt_f32_f16<<<(8388608 + 255) / 256, 256, 0, stream>>>(x, Xh, 8388608);
    cvt_f32_f16<<<(12582912 + 255) / 256, 256, 0, stream>>>(w_qkv, Wqh, 12582912);
    cvt_f32_f16<<<(4194304 + 255) / 256, 256, 0, stream>>>(w_out, Woh, 4194304);

    gemm_bias<true><<<dim3(6144 / 128, 4096 / 128), 256, 0, stream>>>(
        Xh, Wqh, b_qkv, (void*)QKVh, 6144, 2048);

    rope_pack<<<(BB * NHEADS * TT * 64) / 256, 256, 0, stream>>>(QKVh, Qr, Kr, Vt);

    attn_flash<<<dim3(TT / 64, BB * NHEADS), 128, 0, stream>>>(Qr, Kr, Vt, Oh);

    gemm_bias<false><<<dim3(2048 / 128, 4096 / 128), 256, 0, stream>>>(
        Oh, Woh, b_out, (void*)out, 2048, 2048);
}
